// BottleneckBlock_50362786513357
// MI455X (gfx1250) — matrix operand order repaired and checked by reading, not compiled, not run on hardware
//
#include <hip/hip_runtime.h>

// ---------------- problem constants ----------------
constexpr int Bn = 8;      // batch
constexpr int Cc = 256;    // channels
constexpr int Co = 512;    // out channels
constexpr int Cr = 32;     // reduced (q/k) channels
constexpr int Kw = 5;      // conv kernel width
constexpr int Ll = 2048;   // sequence length
constexpr float EPSV  = 1e-5f;
constexpr float SLOPE = 0.2f;

typedef __attribute__((ext_vector_type(16))) __bf16    v16bf;
typedef __attribute__((ext_vector_type(4)))  unsigned  v4u;
typedef __attribute__((ext_vector_type(8)))  float     v8f;

union FragU { v4u q[2]; v16bf bf; };

__device__ __forceinline__ unsigned short f2bf(float f) {
  unsigned u = __float_as_uint(f);
  u += 0x7FFFu + ((u >> 16) & 1u);
  return (unsigned short)(u >> 16);
}

// LDS byte offset of a __shared__ object (low 32 bits of the flat address).
__device__ __forceinline__ unsigned lds_off(const void* p) {
  return (unsigned)(unsigned long long)p;
}

// global -> LDS async copy, 16 bytes (ASYNCcnt-tracked, ISA 15.18.3 op 98)
__device__ __forceinline__ void async_g2l_b128(unsigned lds_byte, const void* g) {
  asm volatile("global_load_async_to_lds_b128 %0, %1, off"
               :: "v"(lds_byte), "v"((unsigned long long)g)
               : "memory");
}

// two aligned 16B LDS reads -> one bf16 fragment (compiler emits ds_load_b128)
__device__ __forceinline__ v16bf frag2(const unsigned short* p0,
                                       const unsigned short* p1) {
  FragU u;
  u.q[0] = *(const v4u*)p0;
  u.q[1] = *(const v4u*)p1;
  return u.bf;
}

// ---------------- fp32 -> bf16 convert (weights) ----------------
__global__ __launch_bounds__(256)
void cvt_f32_bf16(const float* __restrict__ s, unsigned short* __restrict__ d, long n) {
  long i = (long)blockIdx.x * 256 + threadIdx.x;
  if (i < n) d[i] = f2bf(s[i]);
}

// ---------------- x [B][C][L] f32 -> xT [B][L][C] bf16 ----------------
__global__ __launch_bounds__(256)
void cvt_transpose_x(const float* __restrict__ s, unsigned short* __restrict__ d) {
  long i = (long)blockIdx.x * 256 + threadIdx.x;
  if (i >= (long)Bn * Cc * Ll) return;
  long b = i / ((long)Cc * Ll), rem = i % ((long)Cc * Ll);
  long c = rem / Ll, l = rem % Ll;
  d[b * (long)Cc * Ll + l * Cc + c] = f2bf(s[i]);
}

// =====================================================================
// Block-tiled bf16 GEMM: block = WM waves, tile (WM*16) x 128, K-step 32.
// A:  [M,K] row-major (lda).      BT: [N,K] row-major (ldb)  (B transposed!)
// Both async-staged to double-buffered LDS; fragments are contiguous
// 2x b128 LDS reads per lane. Each wave: 8 WMMAs / K-step.
// D = scale*A*B (+bias[m]) (+skip) -> f32 and/or bf16, runtime strides.
// =====================================================================
template <int WM, bool HB, bool HS, bool OF, bool OB>
__global__ __launch_bounds__(WM * 32)
void gemm_tile(const unsigned short* __restrict__ A, long lda, long a_bs,
               const unsigned short* __restrict__ BT, long ldb, long b_bs,
               const float* __restrict__ bias,
               const float* __restrict__ skip, long s_sm, long s_sn, long s_bs,
               float* __restrict__ outf, unsigned short* __restrict__ outb,
               long o_sm, long o_sn, long o_bs,
               int Ktot, float scale) {
  __shared__ unsigned short sA[2][WM * 16 * 32];   // [m][k]
  __shared__ unsigned short sB[2][128 * 32];       // [n][k]

  const int tid  = threadIdx.x;
  const int w    = tid >> 5, lane = tid & 31;
  const int half = lane >> 4, r = lane & 15;
  const int n0   = blockIdx.x * 128;
  const int m0   = blockIdx.y * (WM * 16);
  const long bz  = blockIdx.z;
  A += bz * a_bs; BT += bz * b_bs;

  constexpr int NT  = WM * 32;
  constexpr int NCH = 2 + 512 / NT;   // async chunks per thread per stage

  auto stage = [&](int buf, int k0) {
#pragma unroll
    for (int c = tid; c < WM * 64; c += NT) {        // A: (WM*16)x32
      int row = c >> 2, ko = (c & 3) << 3;
      async_g2l_b128(lds_off(&sA[buf][row * 32 + ko]),
                     A + (long)(m0 + row) * lda + k0 + ko);
    }
#pragma unroll
    for (int c = tid; c < 512; c += NT) {            // BT: 128x32
      int n = c >> 2, ko = (c & 3) << 3;
      async_g2l_b128(lds_off(&sB[buf][n * 32 + ko]),
                     BT + (long)(n0 + n) * ldb + k0 + ko);
    }
  };

  v8f acc[8];
#pragma unroll
  for (int j = 0; j < 8; ++j) acc[j] = {};

  const int steps = Ktot >> 5;
  stage(0, 0);
  for (int i = 0; i < steps; ++i) {
    const int buf = i & 1;
    if (i + 1 < steps) {
      stage(buf ^ 1, (i + 1) << 5);
      asm volatile("s_wait_asynccnt %0" :: "n"(NCH) : "memory");
    } else {
      asm volatile("s_wait_asynccnt 0" ::: "memory");
    }
    __syncthreads();

    // A frag: k = e + (e&8) + 8*half -> two 16B runs at +0 and +32B
    const unsigned short* pa = &sA[buf][(w * 16 + r) * 32 + 8 * half];
    v16bf a = frag2(pa, pa + 16);
#pragma unroll
    for (int j = 0; j < 8; ++j) {
      // B frag: k = e + (e&8) + 8*half, n = j*16+r -> two 16B runs at +0 and +32B
      const unsigned short* pb = &sB[buf][(j * 16 + r) * 32 + 8 * half];
      v16bf b = frag2(pb, pb + 16);
      acc[j] = __builtin_amdgcn_wmma_f32_16x16x32_bf16(false, a, false, b, (short)0,
                                                       acc[j], false, false);
    }
    __syncthreads();
  }

  const float* skp = HS ? skip + bz * s_bs : nullptr;
  float* of = OF ? outf + bz * o_bs : nullptr;
  unsigned short* ob = OB ? outb + bz * o_bs : nullptr;
#pragma unroll
  for (int j = 0; j < 8; ++j) {
    int n = n0 + j * 16 + r;
#pragma unroll
    for (int e = 0; e < 8; ++e) {
      int m = m0 + w * 16 + e + 8 * half;
      float v = acc[j][e] * scale;
      if (HB) v += bias[m];
      if (HS) v += skp[(long)m * s_sm + (long)n * s_sn];
      long oo = (long)m * o_sm + (long)n * o_sn;
      if (OF) of[oo] = v;
      if (OB) ob[oo] = f2bf(v);
    }
  }
}

// =====================================================================
// K=5 conv as implicit-im2col GEMM (WM=4). W async-staged; im2col tile
// gathered from XT [L][C] straight into the [n][k] LDS layout.
// =====================================================================
__global__ __launch_bounds__(128)
void conv5_tile(const unsigned short* __restrict__ W,    // [O, Cc*5]
                const unsigned short* __restrict__ XT,   // [B][Ll][Cc] bf16
                const float* __restrict__ bias,
                float* __restrict__ out, long o_bs) {    // [B][O, Ll] f32
  __shared__ unsigned short sA[2][64 * 32];   // [m][k]
  __shared__ unsigned short sB[2][128 * 32];  // [n][k]

  const int tid  = threadIdx.x;
  const int w    = tid >> 5, lane = tid & 31;
  const int half = lane >> 4, r = lane & 15;
  const int n0   = blockIdx.x * 128;
  const int m0   = blockIdx.y * 64;
  const long bz  = blockIdx.z;
  XT += bz * (long)Cc * Ll;
  out += bz * o_bs;
  constexpr int Ktot = Cc * Kw;  // 1280

  auto stage = [&](int buf, int k0) {
#pragma unroll
    for (int c = tid; c < 256; c += 128) {   // A: 64x32 async
      int row = c >> 2, ko = (c & 3) << 3;
      async_g2l_b128(lds_off(&sA[buf][row * 32 + ko]),
                     W + (long)(m0 + row) * Ktot + k0 + ko);
    }
#pragma unroll
    for (int i = tid; i < 128 * 32; i += 128) {  // im2col gather -> [n][k]
      int n = i >> 5, kq = i & 31;
      int k = k0 + kq, c = k / Kw, kk = k - c * Kw;
      int col = n0 + n + kk - (Kw / 2);
      sB[buf][n * 32 + kq] =
          (col >= 0 && col < Ll) ? XT[(long)col * Cc + c] : (unsigned short)0;
    }
  };

  v8f acc[8];
#pragma unroll
  for (int j = 0; j < 8; ++j) acc[j] = {};

  const int steps = Ktot >> 5;
  stage(0, 0);
  for (int i = 0; i < steps; ++i) {
    const int buf = i & 1;
    if (i + 1 < steps) {
      stage(buf ^ 1, (i + 1) << 5);
      asm volatile("s_wait_asynccnt %0" :: "n"(2) : "memory");
    } else {
      asm volatile("s_wait_asynccnt 0" ::: "memory");
    }
    __syncthreads();

    const unsigned short* pa = &sA[buf][(w * 16 + r) * 32 + 8 * half];
    v16bf a = frag2(pa, pa + 16);
#pragma unroll
    for (int j = 0; j < 8; ++j) {
      const unsigned short* pb = &sB[buf][(j * 16 + r) * 32 + 8 * half];
      v16bf b = frag2(pb, pb + 16);
      acc[j] = __builtin_amdgcn_wmma_f32_16x16x32_bf16(false, a, false, b, (short)0,
                                                       acc[j], false, false);
    }
    __syncthreads();
  }

#pragma unroll
  for (int j = 0; j < 8; ++j) {
    int n = n0 + j * 16 + r;
#pragma unroll
    for (int e = 0; e < 8; ++e) {
      int m = m0 + w * 16 + e + 8 * half;
      out[(long)m * Ll + n] = acc[j][e] + bias[m];
    }
  }
}

// ------ instance norm (over L) + optional residual + leaky ------
// outf: row-major [row][Ll].  outb (optional): b*Cdim*Ll + l*ob_sl + c*ob_sc
__global__ __launch_bounds__(256)
void in_leaky(const float* __restrict__ src, const float* __restrict__ resid,
              float* __restrict__ outf, unsigned short* __restrict__ outb,
              int Cdim, long ob_sc, long ob_sl) {
  __shared__ float s1[256], s2[256];
  long row = blockIdx.x;
  const float* p = src + row * (long)Ll;
  int tid = threadIdx.x;
  float a = 0.f, b = 0.f;
  for (int i = tid; i < Ll; i += 256) { float v = p[i]; a += v; b += v * v; }
  s1[tid] = a; s2[tid] = b; __syncthreads();
  for (int s = 128; s > 0; s >>= 1) {
    if (tid < s) { s1[tid] += s1[tid + s]; s2[tid] += s2[tid + s]; }
    __syncthreads();
  }
  float mean = s1[0] * (1.f / Ll);
  float var  = s2[0] * (1.f / Ll) - mean * mean;
  float inv  = rsqrtf(var + EPSV);
  long bb = row / Cdim, c = row % Cdim;
  for (int i = tid; i < Ll; i += 256) {
    float v = (p[i] - mean) * inv;
    if (resid) v += resid[row * (long)Ll + i];
    v = v >= 0.f ? v : SLOPE * v;
    if (outf) outf[row * (long)Ll + i] = v;
    if (outb) outb[bb * (long)Cdim * Ll + (long)i * ob_sl + c * ob_sc] = f2bf(v);
  }
}

// ---------------- row softmax (2048 cols) -> bf16 probs ----------------
__global__ __launch_bounds__(256)
void softmax_row(float* __restrict__ attn, unsigned short* __restrict__ pb) {
  __shared__ float sm[256];
  long row = blockIdx.x;
  float* p = attn + row * (long)Ll;
  int tid = threadIdx.x;
  float m = -3.4e38f;
  for (int i = tid; i < Ll; i += 256) m = fmaxf(m, p[i]);
  sm[tid] = m; __syncthreads();
  for (int s = 128; s > 0; s >>= 1) {
    if (tid < s) sm[tid] = fmaxf(sm[tid], sm[tid + s]);
    __syncthreads();
  }
  m = sm[0]; __syncthreads();
  float s = 0.f;
  for (int i = tid; i < Ll; i += 256) { float e = __expf(p[i] - m); p[i] = e; s += e; }
  sm[tid] = s; __syncthreads();
  for (int st = 128; st > 0; st >>= 1) {
    if (tid < st) sm[tid] += sm[tid + st];
    __syncthreads();
  }
  float inv = 1.f / sm[0];
  for (int i = tid; i < Ll; i += 256) pb[row * (long)Ll + i] = f2bf(p[i] * inv);
}

// ---------------- launcher ----------------
extern "C" void kernel_launch(void* const* d_in, const int* in_sizes, int n_in,
                              void* d_out, int out_size, void* d_ws, size_t ws_size,
                              hipStream_t stream) {
  (void)in_sizes; (void)n_in; (void)out_size; (void)ws_size;
  const float* x  = (const float*)d_in[0];
  const float* w1 = (const float*)d_in[1];  const float* b1 = (const float*)d_in[2];
  const float* wq = (const float*)d_in[3];  const float* bq = (const float*)d_in[4];
  const float* wk = (const float*)d_in[5];  const float* bk = (const float*)d_in[6];
  const float* wv = (const float*)d_in[7];  const float* bv = (const float*)d_in[8];
  const float* wo = (const float*)d_in[9];  const float* bo = (const float*)d_in[10];
  const float* w2 = (const float*)d_in[11]; const float* b2 = (const float*)d_in[12];
  const float* wr = (const float*)d_in[13]; const float* br = (const float*)d_in[14];
  float* out = (float*)d_out;

  char* wsb = (char*)d_ws;
  size_t off = 0;
  auto alloc = [&](size_t bytes) -> char* {
    char* p = wsb + off;
    off = (off + bytes + 255) & ~(size_t)255;
    return p;
  };
  unsigned short* xT   = (unsigned short*)alloc((size_t)Bn * Ll * Cc * 2); // [l,c]
  unsigned short* w1b  = (unsigned short*)alloc((size_t)Cc * Cc * Kw * 2);
  unsigned short* w2b  = (unsigned short*)alloc((size_t)Co * Cc * Kw * 2);
  unsigned short* wqb  = (unsigned short*)alloc((size_t)Cr * Cc * 2);
  unsigned short* wkb  = (unsigned short*)alloc((size_t)Cr * Cc * 2);
  unsigned short* wvb  = (unsigned short*)alloc((size_t)Cc * Cc * 2);
  unsigned short* wob  = (unsigned short*)alloc((size_t)Cc * Cc * 2);
  unsigned short* wrb  = (unsigned short*)alloc((size_t)Co * Cc * 2);
  float*          t1   = (float*)alloc((size_t)Bn * Cc * Ll * 4);
  float*          h    = (float*)alloc((size_t)Bn * Cc * Ll * 4);
  unsigned short* hT   = (unsigned short*)alloc((size_t)Bn * Ll * Cc * 2); // [l,c]
  unsigned short* qT   = (unsigned short*)alloc((size_t)Bn * Ll * Cr * 2); // [l,cr]
  unsigned short* kT   = (unsigned short*)alloc((size_t)Bn * Ll * Cr * 2); // [l,cr]
  unsigned short* vbf  = (unsigned short*)alloc((size_t)Bn * Cc * Ll * 2); // [c,l]
  float*          attn = (float*)alloc((size_t)Ll * Ll * 4);               // per-batch
  unsigned short* pbf  = (unsigned short*)alloc((size_t)Ll * Ll * 2);      // per-batch
  unsigned short* oT   = (unsigned short*)alloc((size_t)Bn * Ll * Cc * 2); // [l,c]
  unsigned short* h2T  = (unsigned short*)alloc((size_t)Bn * Ll * Cc * 2); // [l,c]
  float*          res  = (float*)alloc((size_t)Bn * Co * Ll * 4);
  float*          t2   = (float*)alloc((size_t)Bn * Co * Ll * 4);

  auto cvt = [&](const float* s, unsigned short* d, long n) {
    cvt_f32_bf16<<<dim3((unsigned)((n + 255) / 256)), dim3(256), 0, stream>>>(s, d, n);
  };
  cvt(w1, w1b, (long)Cc * Cc * Kw);
  cvt(w2, w2b, (long)Co * Cc * Kw);
  cvt(wq, wqb, (long)Cr * Cc);
  cvt(wk, wkb, (long)Cr * Cc);
  cvt(wv, wvb, (long)Cc * Cc);
  cvt(wo, wob, (long)Cc * Cc);
  cvt(wr, wrb, (long)Co * Cc);
  cvt_transpose_x<<<dim3((Bn * Cc * Ll + 255) / 256), dim3(256), 0, stream>>>(x, xT);

  const long CL = (long)Cc * Ll, CoL = (long)Co * Ll, CrL = (long)Cr * Ll;

  // conv1 -> t1 ; IN + leaky -> h (f32 [c,l]) + hT (bf16 [l,c])
  conv5_tile<<<dim3(Ll / 128, Cc / 64, Bn), dim3(128), 0, stream>>>(
      w1b, xT, b1, t1, CL);
  in_leaky<<<dim3(Bn * Cc), dim3(256), 0, stream>>>(t1, nullptr, h, hT, Cc, 1, Cc);

  // q = wq*h+bq -> qT [L,Cr]  (transposed store; WM=2 since M=32)
  gemm_tile<2, true, false, false, true><<<dim3(Ll / 128, 1, Bn), dim3(64), 0, stream>>>(
      wqb, Cc, 0, hT, Cc, CL, bq, nullptr, 0, 0, 0,
      nullptr, qT, 1, Cr, CrL, Cc, 1.0f);
  // k = wk*h+bk -> kT [L,Cr]
  gemm_tile<2, true, false, false, true><<<dim3(Ll / 128, 1, Bn), dim3(64), 0, stream>>>(
      wkb, Cc, 0, hT, Cc, CL, bk, nullptr, 0, 0, 0,
      nullptr, kT, 1, Cr, CrL, Cc, 1.0f);
  // v = wv*h+bv -> v [C,L] row-major
  gemm_tile<4, true, false, false, true><<<dim3(Ll / 128, Cc / 64, Bn), dim3(128), 0, stream>>>(
      wvb, Cc, 0, hT, Cc, CL, bv, nullptr, 0, 0, 0,
      nullptr, vbf, Ll, 1, CL, Cc, 1.0f);
  // residual = wr*x+br -> f32 [Co,L]
  gemm_tile<4, true, false, true, false><<<dim3(Ll / 128, Co / 64, Bn), dim3(128), 0, stream>>>(
      wrb, Cc, 0, xT, Cc, CL, br, nullptr, 0, 0, 0,
      res, nullptr, Ll, 1, CoL, Cc, 1.0f);

  // attention per batch (attn/pbf scratch reused; stream order serializes)
  const float qscale = 0.17677669529663687f;  // Cr^-0.5
  for (int bi = 0; bi < Bn; ++bi) {
    // attn[l,m] = scale * qT[l,:].kT[m,:]   (A=qT lda=Cr, BT=kT ldb=Cr)
    gemm_tile<4, false, false, true, false><<<dim3(Ll / 128, Ll / 64, 1), dim3(128), 0, stream>>>(
        qT + bi * CrL, Cr, 0, kT + bi * CrL, Cr, 0,
        nullptr, nullptr, 0, 0, 0,
        attn, nullptr, Ll, 1, 0, Cr, qscale);
    softmax_row<<<dim3(Ll), dim3(256), 0, stream>>>(attn, pbf);
    // oT[l,c] = sum_m p[l,m] v[c,m]   (A=p lda=Ll, BT=v ldb=Ll) -> [l,c]
    gemm_tile<4, false, false, false, true><<<dim3(Cc / 128, Ll / 64, 1), dim3(128), 0, stream>>>(
        pbf, Ll, 0, vbf + bi * CL, Ll, 0,
        nullptr, nullptr, 0, 0, 0,
        nullptr, oT + bi * CL, Cc, 1, 0, Ll, 1.0f);
  }

  // h2 = wo*o + bo + h -> h2T [L,C]  (BT = oT ldb=Cc; skip=h [c,l])
  gemm_tile<4, true, true, false, true><<<dim3(Ll / 128, Cc / 64, Bn), dim3(128), 0, stream>>>(
      wob, Cc, 0, oT, Cc, CL, bo, h, Ll, 1, CL,
      nullptr, h2T, 1, Cc, CL, Cc, 1.0f);

  // conv2 -> t2 ; IN + residual + leaky -> out
  conv5_tile<<<dim3(Ll / 128, Co / 64, Bn), dim3(128), 0, stream>>>(
      w2b, h2T, b2, t2, CoL);
  in_leaky<<<dim3(Bn * Co), dim3(256), 0, stream>>>(t2, res, out, nullptr, Co, 0, 0);
}
